// GINet_45002667327983
// MI455X (gfx1250) — matrix operand order repaired and checked by reading, not compiled, not run on hardware
//
#include <hip/hip_runtime.h>
#include <hip/hip_bf16.h>
#include <math.h>

// Problem constants (from reference)
#define NNODE 20000
#define NEDGE 320000
#define EMB   300
#define NGB   256      // graphs in batch
#define NLAYER 5
// Padded GEMM dims
#define MP    20096    // 20000 padded to multiple of 128
#define EMBP  320      // 300 -> 320 (10 k-steps of 32)
#define HIDP  640      // 600 -> 640 (20 k-steps of 32)

typedef _Float16     v16h __attribute__((ext_vector_type(16)));
typedef float        v8f  __attribute__((ext_vector_type(8)));
typedef unsigned int v4u  __attribute__((ext_vector_type(4)));

union Frag  { v4u u[2]; v16h h; };
union Pack2 { _Float16 h[2]; unsigned int u; };

// ---------------------------------------------------------------- utilities
__global__ void zero_kernel(float* __restrict__ p, long n) {
    long i = (long)blockIdx.x * blockDim.x + threadIdx.x;
    long step = (long)gridDim.x * blockDim.x;
    for (; i < n; i += step) p[i] = 0.0f;
}

// ------------------------------------------------- node embedding sum -> h
__global__ void node_init_kernel(const int* __restrict__ x,
                                 const float* __restrict__ node_emb,
                                 float* __restrict__ h) {
    const int off[9] = {0, 119, 127, 139, 154, 164, 170, 177, 180};
    long total = (long)NNODE * EMB;
    long i = (long)blockIdx.x * blockDim.x + threadIdx.x;
    long step = (long)gridDim.x * blockDim.x;
    for (; i < total; i += step) {
        int n = (int)(i / EMB);
        int c = (int)(i % EMB);
        float s = 0.0f;
        #pragma unroll
        for (int t = 0; t < 9; ++t)
            s += node_emb[(long)(x[n * 9 + t] + off[t]) * EMB + c];
        h[i] = s;
    }
}

// ---------------------------------------- edge message + scatter (atomics)
__global__ void edge_scatter_kernel(const int* __restrict__ ei,
                                    const int* __restrict__ attr,
                                    const float* __restrict__ eemb,  // [31, EMB] for this layer
                                    const float* __restrict__ h,
                                    float* __restrict__ agg) {
    long total = (long)NEDGE * EMB;
    long i = (long)blockIdx.x * blockDim.x + threadIdx.x;
    long step = (long)gridDim.x * blockDim.x;
    for (; i < total; i += step) {
        int e = (int)(i / EMB);
        int c = (int)(i % EMB);
        int r   = ei[e];
        int col = ei[NEDGE + e];
        int a0 = attr[e * 3 + 0];         // + EDGE_OFF[0]=0
        int a1 = attr[e * 3 + 1] + 23;    // + EDGE_OFF[1]
        int a2 = attr[e * 3 + 2] + 29;    // + EDGE_OFF[2]
        float ee = eemb[(long)a0 * EMB + c] + eemb[(long)a1 * EMB + c] + eemb[(long)a2 * EMB + c];
        atomicAdd(&agg[(long)col * EMB + c], h[(long)r * EMB + c] + ee);
    }
}

// self loops: attr [22,0,0] -> rows 22, 23, 29 of this layer's table (constant vec)
__global__ void self_loop_kernel(const float* __restrict__ eemb,
                                 const float* __restrict__ h,
                                 float* __restrict__ agg) {
    long total = (long)NNODE * EMB;
    long i = (long)blockIdx.x * blockDim.x + threadIdx.x;
    long step = (long)gridDim.x * blockDim.x;
    for (; i < total; i += step) {
        int c = (int)(i % EMB);
        float ee = eemb[22 * EMB + c] + eemb[23 * EMB + c] + eemb[29 * EMB + c];
        agg[i] += h[i] + ee;
    }
}

// -------------------- f32 [NNODE, EMB] -> f16-pair uints [MP, EMBP/2], zero pad
__global__ void conv_a_kernel(const float* __restrict__ src,
                              unsigned int* __restrict__ A16) {
    long total = (long)MP * (EMBP / 2);
    long i = (long)blockIdx.x * blockDim.x + threadIdx.x;
    long step = (long)gridDim.x * blockDim.x;
    for (; i < total; i += step) {
        int m  = (int)(i / (EMBP / 2));
        int c0 = 2 * (int)(i % (EMBP / 2));
        Pack2 p;
        float f0 = (m < NNODE && c0 < EMB)     ? src[(long)m * EMB + c0]     : 0.0f;
        float f1 = (m < NNODE && c0 + 1 < EMB) ? src[(long)m * EMB + c0 + 1] : 0.0f;
        p.h[0] = (_Float16)f0;
        p.h[1] = (_Float16)f1;
        A16[i] = p.u;
    }
}

// ---------- repack W [Kreal, Nreal] f32 -> fragment-native f16 B buffer
// Layout: Bf[((kt*NcolsP + n)*2 + half)*8 + i] packs B[k,n],B[k+1,n]
// with k = kt*32 + (i/4)*16 + half*8 + (i%4)*2   (ISA 7.12.2 dense 16-bit B layout)
__global__ void prep_b_kernel(const float* __restrict__ W,
                              unsigned int* __restrict__ Bf,
                              int Kp, int NcolsP, int Kreal, int Nreal) {
    long total = (long)(Kp >> 5) * NcolsP * 16;
    long idx = (long)blockIdx.x * blockDim.x + threadIdx.x;
    long step = (long)gridDim.x * blockDim.x;
    for (; idx < total; idx += step) {
        int i  = (int)(idx & 7);
        int hh = (int)((idx >> 3) & 1);
        long rest = idx >> 4;
        int n  = (int)(rest % NcolsP);
        int kt = (int)(rest / NcolsP);
        int k = kt * 32 + (i >> 2) * 16 + hh * 8 + (i & 3) * 2;
        float w0 = (k < Kreal && n < Nreal)     ? W[(long)k * Nreal + n]       : 0.0f;
        float w1 = (k + 1 < Kreal && n < Nreal) ? W[(long)(k + 1) * Nreal + n] : 0.0f;
        Pack2 p;
        p.h[0] = (_Float16)w0;
        p.h[1] = (_Float16)w1;
        Bf[idx] = p.u;
    }
}

// ----------------------------------------------------------- WMMA GEMM
// A: [Mp x Kp/2] uints (row-major f16 pairs). Bf: fragment-native (above).
// Each wave computes a 32(M) x 64(N) tile: 2x4 wmma accumulators.
// block = 128 threads = 4 waves; wavesN in {1,2}: block tile (128x64)/(64x128).
// mode 0: out = f16 relu(acc + bias), full [Mp x NcolsP]
// mode 1: out = f32 (acc + bias) stored compact [Mreal x EMB], guarded
__global__ __launch_bounds__(128)
void wmma_gemm_kernel(const unsigned int* __restrict__ A,
                      const unsigned int* __restrict__ Bf,
                      const float* __restrict__ bias,
                      _Float16* __restrict__ outH,
                      float* __restrict__ outF,
                      int Kp, int NcolsP, int wavesN,
                      int mode, int Mreal, int biasN) {
    const int lane = threadIdx.x & 31;
    const int wave = threadIdx.x >> 5;
    const int half = lane >> 4;
    const int lm   = lane & 15;
    const int wavesM = 4 / wavesN;
    const int wm = wave / wavesN;
    const int wn = wave % wavesN;
    const int mBase = (blockIdx.x * wavesM + wm) * 32;
    const int nBase = (blockIdx.y * wavesN + wn) * 64;

    v8f acc[8] = {};
    const int K2 = Kp >> 1;
    const v4u* A4 = (const v4u*)A;
    const v4u* B4 = (const v4u*)Bf;

    for (int kt = 0; kt < (Kp >> 5); ++kt) {
        Frag af[2];
        #pragma unroll
        for (int i = 0; i < 2; ++i) {
            int m = mBase + i * 16 + lm;
            // A frag: u[j] holds k = kt*32 + (j/4)*16 + half*8 + (j%4)*2 -> two
            // runs of 4 consecutive uints at k/2 = kt*16 + half*4 (+8 for j>=4)
            int base = m * K2 + kt * 16 + half * 4;   // uint index, %4 == 0
            af[i].u[0] = A4[(base >> 2)];
            af[i].u[1] = A4[(base >> 2) + 2];
        }
        Frag bfrag[4];
        #pragma unroll
        for (int j = 0; j < 4; ++j) {
            int n = nBase + j * 16 + lm;
            int base = ((kt * NcolsP + n) * 2 + half) * 8;  // uint index, %8 == 0
            bfrag[j].u[0] = B4[(base >> 2)];
            bfrag[j].u[1] = B4[(base >> 2) + 1];
        }
        #pragma unroll
        for (int i = 0; i < 2; ++i)
            #pragma unroll
            for (int j = 0; j < 4; ++j)
                acc[i * 4 + j] = __builtin_amdgcn_wmma_f32_16x16x32_f16(
                    false, af[i].h, false, bfrag[j].h,
                    (short)0, acc[i * 4 + j], false, false);
    }

    #pragma unroll
    for (int j = 0; j < 4; ++j) {
        int n = nBase + j * 16 + lm;
        float bv = (n < biasN) ? bias[n] : 0.0f;
        #pragma unroll
        for (int i = 0; i < 2; ++i) {
            #pragma unroll
            for (int r = 0; r < 8; ++r) {
                int m = mBase + i * 16 + half * 8 + r;   // C/D layout (ISA 7.12.2)
                float v = acc[i * 4 + j][r] + bv;
                if (mode == 0) {
                    v = v > 0.0f ? v : 0.0f;
                    outH[(long)m * NcolsP + n] = (_Float16)v;
                } else {
                    if (m < Mreal && n < EMB) outF[(long)m * EMB + n] = v;
                }
            }
        }
    }
}

// -------------------------------------------------- batchnorm (train stats)
__global__ __launch_bounds__(256)
void bn_kernel(const float* __restrict__ src, float* __restrict__ dst,
               const float* __restrict__ gamma, const float* __restrict__ beta,
               int relu) {
    int c = blockIdx.x;            // one column per block, c in [0, EMB)
    __shared__ float s1[256], s2[256];
    __shared__ float mu_s, rs_s;
    int tid = threadIdx.x;
    float sum = 0.0f, sq = 0.0f;
    for (int n = tid; n < NNODE; n += 256) {
        float v = src[(long)n * EMB + c];
        sum += v; sq += v * v;
    }
    s1[tid] = sum; s2[tid] = sq;
    __syncthreads();
    for (int s = 128; s > 0; s >>= 1) {
        if (tid < s) { s1[tid] += s1[tid + s]; s2[tid] += s2[tid + s]; }
        __syncthreads();
    }
    if (tid == 0) {
        float mu = s1[0] / (float)NNODE;
        float var = s2[0] / (float)NNODE - mu * mu;
        mu_s = mu;
        rs_s = rsqrtf(var + 1e-5f);
    }
    __syncthreads();
    float mu = mu_s, rs = rs_s, g = gamma[c], bb = beta[c];
    for (int n = tid; n < NNODE; n += 256) {
        float v = (src[(long)n * EMB + c] - mu) * rs * g + bb;
        if (relu) v = fmaxf(v, 0.0f);
        dst[(long)n * EMB + c] = v;
    }
}

// ------------------------------------------------------------- mean pooling
__global__ void pool_count_kernel(const int* __restrict__ xb, float* __restrict__ counts) {
    long i = (long)blockIdx.x * blockDim.x + threadIdx.x;
    if (i < NNODE) atomicAdd(&counts[xb[i]], 1.0f);
}

__global__ void pool_sum_kernel(const float* __restrict__ h, const int* __restrict__ xb,
                                float* __restrict__ hg) {
    long total = (long)NNODE * EMB;
    long i = (long)blockIdx.x * blockDim.x + threadIdx.x;
    long step = (long)gridDim.x * blockDim.x;
    for (; i < total; i += step) {
        int n = (int)(i / EMB);
        int c = (int)(i % EMB);
        atomicAdd(&hg[(long)xb[n] * EMB + c], h[i]);
    }
}

__global__ void pool_div_kernel(float* __restrict__ hg, const float* __restrict__ counts) {
    long i = (long)blockIdx.x * blockDim.x + threadIdx.x;
    if (i < (long)NGB * EMB) {
        int b = (int)(i / EMB);
        hg[i] /= fmaxf(counts[b], 1.0f);
    }
}

// -------------------------------------------- small dense layers (f32 head)
__global__ void dense_kernel(const float* __restrict__ X, const float* __restrict__ W,
                             const float* __restrict__ bvec, float* __restrict__ Y,
                             int rows, int K, int O, int softplus) {
    long total = (long)rows * O;
    long i = (long)blockIdx.x * blockDim.x + threadIdx.x;
    long step = (long)gridDim.x * blockDim.x;
    for (; i < total; i += step) {
        int r = (int)(i / O);
        int o = (int)(i % O);
        float s = bvec[o];
        for (int k = 0; k < K; ++k) s += X[(long)r * K + k] * W[(long)k * O + o];
        if (softplus) s = (s > 20.0f) ? s : log1pf(expf(s));
        Y[i] = s;
    }
}

// ---------------------------------------------------------------- launcher
static inline int cdiv(long a, long b) { return (int)((a + b - 1) / b); }

extern "C" void kernel_launch(void* const* d_in, const int* in_sizes, int n_in,
                              void* d_out, int out_size, void* d_ws, size_t ws_size,
                              hipStream_t stream) {
    const int*   x        = (const int*)d_in[0];
    const int*   ei       = (const int*)d_in[1];
    const int*   attr     = (const int*)d_in[2];
    const int*   xb       = (const int*)d_in[3];
    const float* node_emb = (const float*)d_in[4];
    const float* edge_emb = (const float*)d_in[5];
    const float* W1       = (const float*)d_in[6];
    const float* b1       = (const float*)d_in[7];
    const float* W2       = (const float*)d_in[8];
    const float* b2       = (const float*)d_in[9];
    const float* gamma    = (const float*)d_in[10];
    const float* beta     = (const float*)d_in[11];
    const float* feat_w   = (const float*)d_in[12];
    const float* feat_b   = (const float*)d_in[13];
    const float* pw1      = (const float*)d_in[14];
    const float* pb1      = (const float*)d_in[15];
    const float* pw2      = (const float*)d_in[16];
    const float* pb2      = (const float*)d_in[17];
    const float* pw3      = (const float*)d_in[18];
    const float* pb3      = (const float*)d_in[19];

    char* ws = (char*)d_ws;
    size_t off = 0;
    float*        h      = (float*)(ws + off);        off += (size_t)NNODE * EMB * 4;        // 24.0 MB
    float*        agg    = (float*)(ws + off);        off += (size_t)NNODE * EMB * 4;        // 24.0 MB
    unsigned int* A16    = (unsigned int*)(ws + off); off += (size_t)MP * EMBP * 2;          // 12.9 MB
    _Float16*     Z16    = (_Float16*)(ws + off);     off += (size_t)MP * HIDP * 2;          // 25.7 MB
    unsigned int* W1f    = (unsigned int*)(ws + off); off += (size_t)EMBP * HIDP * 2;        // 0.4 MB
    unsigned int* W2f    = (unsigned int*)(ws + off); off += (size_t)HIDP * EMBP * 2;        // 0.4 MB
    float*        hg     = (float*)(ws + off);        off += (size_t)NGB * EMB * 4;
    float*        counts = (float*)(ws + off);        off += (size_t)NGB * 4;
    float*        feat   = (float*)(ws + off);        off += (size_t)NGB * 512 * 4;
    float*        p1     = (float*)(ws + off);        off += (size_t)NGB * 256 * 4;
    float*        p2     = (float*)(ws + off);        off += (size_t)NGB * 256 * 4;

    const int T = 256;

    // h0 = sum of 9 node-feature embeddings
    node_init_kernel<<<cdiv((long)NNODE * EMB, T), T, 0, stream>>>(x, node_emb, h);

    for (int l = 0; l < NLAYER; ++l) {
        const float* eemb = edge_emb + (long)l * 31 * EMB;

        zero_kernel<<<cdiv((long)NNODE * EMB, T), T, 0, stream>>>(agg, (long)NNODE * EMB);
        edge_scatter_kernel<<<cdiv((long)NEDGE * EMB, T), T, 0, stream>>>(ei, attr, eemb, h, agg);
        self_loop_kernel<<<cdiv((long)NNODE * EMB, T), T, 0, stream>>>(eemb, h, agg);

        // agg -> f16 padded A
        conv_a_kernel<<<cdiv((long)MP * (EMBP / 2), T), T, 0, stream>>>(agg, A16);

        // GEMM1: [MP,320] x [320,640] -> relu f16 [MP,640]
        prep_b_kernel<<<cdiv((long)(EMBP / 32) * HIDP * 16, T), T, 0, stream>>>(
            W1 + (long)l * EMB * 600, W1f, EMBP, HIDP, EMB, 600);
        {
            dim3 grid(MP / 64, HIDP / 128);
            wmma_gemm_kernel<<<grid, 128, 0, stream>>>(
                A16, W1f, b1 + (long)l * 600, Z16, (float*)nullptr,
                EMBP, HIDP, /*wavesN=*/2, /*mode=*/0, MP, 600);
        }

        // GEMM2: [MP,640] x [640,320] -> f32 [NNODE,300] into agg (pre-BN)
        prep_b_kernel<<<cdiv((long)(HIDP / 32) * EMBP * 16, T), T, 0, stream>>>(
            W2 + (long)l * 600 * EMB, W2f, HIDP, EMBP, 600, EMB);
        {
            dim3 grid(MP / 128, EMBP / 64);
            wmma_gemm_kernel<<<grid, 128, 0, stream>>>(
                (const unsigned int*)Z16, W2f, b2 + (long)l * EMB,
                (_Float16*)nullptr, agg,
                HIDP, EMBP, /*wavesN=*/1, /*mode=*/1, NNODE, EMB);
        }

        // BatchNorm (+ ReLU except last layer): agg -> h
        bn_kernel<<<EMB, 256, 0, stream>>>(agg, h, gamma + (long)l * EMB,
                                           beta + (long)l * EMB, (l < NLAYER - 1) ? 1 : 0);
    }

    // mean pooling per graph (hg and counts are contiguous in ws)
    zero_kernel<<<cdiv((long)NGB * EMB + NGB, T), T, 0, stream>>>(hg, (long)NGB * EMB + NGB);
    pool_count_kernel<<<cdiv(NNODE, T), T, 0, stream>>>(xb, counts);
    pool_sum_kernel<<<cdiv((long)NNODE * EMB, T), T, 0, stream>>>(h, xb, hg);
    pool_div_kernel<<<cdiv((long)NGB * EMB, T), T, 0, stream>>>(hg, counts);

    // prediction head (f32, tiny)
    dense_kernel<<<cdiv((long)NGB * 512, T), T, 0, stream>>>(hg, feat_w, feat_b, feat, NGB, EMB, 512, 0);
    dense_kernel<<<cdiv((long)NGB * 256, T), T, 0, stream>>>(feat, pw1, pb1, p1, NGB, 512, 256, 1);
    dense_kernel<<<cdiv((long)NGB * 256, T), T, 0, stream>>>(p1, pw2, pb2, p2, NGB, 256, 256, 1);
    dense_kernel<<<cdiv((long)NGB * 2, T), T, 0, stream>>>(p2, pw3, pb3, (float*)d_out, NGB, 256, 2, 0);
}
